// EM_Word_Attention_72919954751621
// MI455X (gfx1250) — matrix operand order repaired and checked by reading, not compiled, not run on hardware
//
#include <hip/hip_runtime.h>
#include <hip/hip_bf16.h>
#include <math.h>

// Problem dims (fixed by reference)
#define B_DIM 32
#define E_DIM 512
#define W_N   2048
#define D_DIM 1024

typedef __bf16 v16bf __attribute__((ext_vector_type(16)));
typedef float  v8f   __attribute__((ext_vector_type(8)));

union V16 {
  uint4          q[2];
  unsigned short s[16];
  v16bf          v;
};

__device__ __forceinline__ unsigned short f2bf(float f) {
  unsigned u = __float_as_uint(f);
  unsigned r = u + 0x7FFFu + ((u >> 16) & 1u);   // round-to-nearest-even
  return (unsigned short)(r >> 16);
}

// ---------------------------------------------------------------------------
// Stage 0a: fp32 -> bf16 bulk convert (float4 -> ushort4)
// ---------------------------------------------------------------------------
__global__ void cvt_f32_to_bf16(const float* __restrict__ src,
                                unsigned short* __restrict__ dst, int n4) {
  int i = blockIdx.x * blockDim.x + threadIdx.x;
  if (i >= n4) return;
  float4 v = ((const float4*)src)[i];
  ushort4 o;
  o.x = f2bf(v.x); o.y = f2bf(v.y); o.z = f2bf(v.z); o.w = f2bf(v.w);
  ((ushort4*)dst)[i] = o;
}

// ---------------------------------------------------------------------------
// Stage 0b: words [B][Wn][D] f32  ->  WD16T [B][D][Wn] bf16 (transposed copy)
// 32x32 LDS tile, 32x8 threads.
// ---------------------------------------------------------------------------
__global__ __launch_bounds__(256) void transpose_cvt_kernel(
    const float* __restrict__ src, unsigned short* __restrict__ dst) {
  __shared__ unsigned short tile[32][33];   // +1 pad kills bank conflicts
  const int b  = blockIdx.z;
  const int w0 = blockIdx.x * 32;   // dim block
  const int n0 = blockIdx.y * 32;   // word block
  const int tx = threadIdx.x, ty = threadIdx.y;
#pragma unroll
  for (int i = 0; i < 4; ++i) {
    const int n = n0 + ty + i * 8;
    tile[ty + i * 8][tx] = f2bf(src[((size_t)b * W_N + n) * D_DIM + w0 + tx]);
  }
  __syncthreads();
#pragma unroll
  for (int i = 0; i < 4; ++i) {
    const int w = w0 + ty + i * 8;
    dst[((size_t)b * D_DIM + w) * W_N + n0 + tx] = tile[tx][ty + i * 8];
  }
}

// ---------------------------------------------------------------------------
// Stage 1: proj = tanh(ems @ W^T + b). Double-buffered bf16 WMMA GEMM.
// Block = 16 rows x 1024 cols, 8 waves x 128 cols (8 C tiles each).
// ---------------------------------------------------------------------------
__global__ __launch_bounds__(256) void proj_tanh_kernel(
    const unsigned short* __restrict__ E16,
    const unsigned short* __restrict__ W16,
    const float* __restrict__ bias,
    unsigned short* __restrict__ P16) {
  const int lane = threadIdx.x & 31;
  const int wave = threadIdx.x >> 5;
  const int h    = lane >> 4;
  const int l16  = lane & 15;
  const int rowBase = blockIdx.x * 16;
  const int colBase = wave * 128;

  v8f acc[8];
#pragma unroll
  for (int t = 0; t < 8; ++t)
#pragma unroll
    for (int r = 0; r < 8; ++r) acc[t][r] = 0.f;

  const unsigned short* aRow = E16 + (size_t)(rowBase + l16) * D_DIM;
  const unsigned short* bRow = W16 + (size_t)(colBase + l16) * D_DIM + h * 16;

  auto LD = [&](int k0, V16& a, V16 bset[8]) {
    a.q[0] = *(const uint4*)(aRow + k0 + h * 16);
    a.q[1] = *(const uint4*)(aRow + k0 + h * 16 + 8);
#pragma unroll
    for (int t = 0; t < 8; ++t) {
      const unsigned short* bp = bRow + (size_t)t * 16 * D_DIM + k0;
      bset[t].q[0] = *(const uint4*)(bp);
      bset[t].q[1] = *(const uint4*)(bp + 8);
    }
  };

  V16 a0, a1, b0[8], b1[8];
  LD(0, a0, b0);
  for (int k0 = 0; k0 < D_DIM; k0 += 64) {
    LD(k0 + 32, a1, b1);                       // prefetch next 32-K slab
#pragma unroll
    for (int t = 0; t < 8; ++t)
      acc[t] = __builtin_amdgcn_wmma_f32_16x16x32_bf16(
          false, a0.v, false, b0[t].v, (short)0, acc[t], false, false);
    if (k0 + 64 < D_DIM) LD(k0 + 64, a0, b0);  // prefetch slab after that
#pragma unroll
    for (int t = 0; t < 8; ++t)
      acc[t] = __builtin_amdgcn_wmma_f32_16x16x32_bf16(
          false, a1.v, false, b1[t].v, (short)0, acc[t], false, false);
  }
#pragma unroll
  for (int t = 0; t < 8; ++t) {
    const int col = colBase + t * 16 + l16;
    const float bv = bias[col];
#pragma unroll
    for (int r = 0; r < 8; ++r) {
      const int row = rowBase + r + 8 * h;
      P16[(size_t)row * D_DIM + col] = f2bf(tanhf(acc[t][r] + bv));
    }
  }
}

// ---------------------------------------------------------------------------
// Stage 2: flash-attention. Block = (batch b, 16 E-rows). 8 waves.
//  - scores: waves split K (128 each), split-K reduced via ds_add_f32
//  - online masked softmax by wave 0, P -> LDS in WMMA A-layout (bf16)
//  - PV: waves split ctx columns (128 each); B from pre-transposed WD16T
// All fragments batch-loaded (one clause + one wait per 8-WMMA burst).
// ---------------------------------------------------------------------------
__global__ __launch_bounds__(256) void attn_kernel(
    const unsigned short* __restrict__ P16,
    const unsigned short* __restrict__ WD16,
    const unsigned short* __restrict__ WD16T,
    const unsigned char* __restrict__ mask,
    float* __restrict__ out) {
  __shared__ __align__(16) float          S[16][32];
  __shared__ __align__(16) unsigned short Pl[16][32];
  __shared__ float m_run[16], l_run[16], resc[16];

  const int lane = threadIdx.x & 31;
  const int wave = threadIdx.x >> 5;
  const int h    = lane >> 4;
  const int l16  = lane & 15;
  const int b    = blockIdx.x >> 5;
  const int e0   = (blockIdx.x & 31) * 16;
  const int kBase   = wave * 128;
  const int colBase = wave * 128;
  const size_t qRow0 = (size_t)(b * E_DIM + e0);
  const size_t wRow0 = (size_t)b * W_N;
  const size_t tRow0 = (size_t)b * D_DIM;
  const float NEG_INF = -__builtin_inff();

  v8f acc[8];
#pragma unroll
  for (int t = 0; t < 8; ++t)
#pragma unroll
    for (int r = 0; r < 8; ++r) acc[t][r] = 0.f;

  if (threadIdx.x < 16) { m_run[threadIdx.x] = NEG_INF; l_run[threadIdx.x] = 0.f; }
  __syncthreads();

  for (int n0 = 0; n0 < W_N; n0 += 32) {
    // zero split-K buffer (512 floats / 256 threads)
    float* sf = &S[0][0];
    sf[threadIdx.x]       = 0.f;
    sf[threadIdx.x + 256] = 0.f;

    // hint next n-block into cache while this one computes
    if (n0 + 32 < W_N) {
      __builtin_prefetch(WD16 + (wRow0 + n0 + 32 + lane) * D_DIM + kBase, 0, 0);
      __builtin_prefetch(WD16T + (tRow0 + colBase + lane) * W_N + n0 + 32, 0, 0);
    }

    // ---- batch-load all scores fragments (global, no LDS dependence) ----
    V16 aa[4], bb[8];
#pragma unroll
    for (int kk = 0; kk < 4; ++kk) {
      const int k0 = kBase + kk * 32;
      const unsigned short* ap = P16 + (qRow0 + l16) * D_DIM + k0;
      aa[kk].q[0] = *(const uint4*)(ap + h * 16);
      aa[kk].q[1] = *(const uint4*)(ap + h * 16 + 8);
#pragma unroll
      for (int tn = 0; tn < 2; ++tn) {
        const unsigned short* bp =
            WD16 + (wRow0 + n0 + tn * 16 + l16) * D_DIM + k0 + h * 16;
        bb[kk * 2 + tn].q[0] = *(const uint4*)(bp);
        bb[kk * 2 + tn].q[1] = *(const uint4*)(bp + 8);
      }
    }
    __syncthreads();   // S zeroed everywhere before atomics

    v8f sc[2];
#pragma unroll
    for (int t = 0; t < 2; ++t)
#pragma unroll
      for (int r = 0; r < 8; ++r) sc[t][r] = 0.f;
#pragma unroll
    for (int kk = 0; kk < 4; ++kk)
#pragma unroll
      for (int tn = 0; tn < 2; ++tn)
        sc[tn] = __builtin_amdgcn_wmma_f32_16x16x32_bf16(
            false, aa[kk].v, false, bb[kk * 2 + tn].v, (short)0, sc[tn],
            false, false);

#pragma unroll
    for (int tn = 0; tn < 2; ++tn)
#pragma unroll
      for (int r = 0; r < 8; ++r)
        atomicAdd(&S[r + 8 * h][tn * 16 + l16], sc[tn][r]);
    __syncthreads();

    // ---- PV B fragments from transposed words: contiguous b128 loads.
    //      Issued *before* softmax so they fly during wave0's work. ----
    V16 pb[8];
#pragma unroll
    for (int t = 0; t < 8; ++t) {
      const unsigned short* wp =
          WD16T + (tRow0 + colBase + t * 16 + l16) * W_N + n0 + h * 16;
      pb[t].q[0] = *(const uint4*)(wp);
      pb[t].q[1] = *(const uint4*)(wp + 8);
    }

    // ---- online masked softmax (wave 0) ----
    if (wave == 0) {
      const int r = l16;
      union { uint4 q; unsigned char c[16]; } mu;
      mu.q = *(const uint4*)(mask + (qRow0 + r) * W_N + n0 + h * 16);
      float vals[16];
      float mloc = NEG_INF;
#pragma unroll
      for (int j = 0; j < 16; ++j) {
        float s = S[r][h * 16 + j];
        if (mu.c[j]) s = NEG_INF;   // mask==True -> exclude
        vals[j] = s;
        mloc = fmaxf(mloc, s);
      }
      const float moth = __shfl_xor(mloc, 16, 32);
      const float mblk = fmaxf(mloc, moth);
      const float mold = m_run[r];
      const float mnew = fmaxf(mold, mblk);
      float sum = 0.f;
      union { uint4 q[2]; unsigned short s[16]; } pu;
#pragma unroll
      for (int j = 0; j < 16; ++j) {
        float p = mu.c[j] ? 0.f : __expf(vals[j] - mnew);
        pu.s[j] = f2bf(p);
        sum += p;
      }
      sum += __shfl_xor(sum, 16, 32);
      uint4* pd = (uint4*)&Pl[r][h * 16];
      pd[0] = pu.q[0];
      pd[1] = pu.q[1];
      if (h == 0) {
        const float rs = (mnew == NEG_INF) ? 0.f : __expf(mold - mnew);
        resc[r]  = rs;
        l_run[r] = l_run[r] * rs + sum;
        m_run[r] = mnew;
      }
    }
    __syncthreads();

    // ---- rescale accumulators, then ctx += P @ words_tile ----
    float rs[8];
#pragma unroll
    for (int r = 0; r < 8; ++r) rs[r] = resc[r + 8 * h];
#pragma unroll
    for (int t = 0; t < 8; ++t)
#pragma unroll
      for (int r = 0; r < 8; ++r) acc[t][r] *= rs[r];

    V16 pa;  // A = P 16x32 from LDS (A-layout chunks)
    const unsigned short* pp = &Pl[l16][0];
    pa.q[0] = *(const uint4*)(pp + h * 16);
    pa.q[1] = *(const uint4*)(pp + h * 16 + 8);
#pragma unroll
    for (int t = 0; t < 8; ++t)
      acc[t] = __builtin_amdgcn_wmma_f32_16x16x32_bf16(
          false, pa.v, false, pb[t].v, (short)0, acc[t], false, false);
  }

  // ---- epilogue: ctx / l, fp32 out ----
  float inv[8];
#pragma unroll
  for (int r = 0; r < 8; ++r) inv[r] = 1.f / l_run[r + 8 * h];
#pragma unroll
  for (int t = 0; t < 8; ++t)
#pragma unroll
    for (int r = 0; r < 8; ++r)
      out[(qRow0 + r + 8 * h) * D_DIM + colBase + t * 16 + l16] =
          acc[t][r] * inv[r];
}

// ---------------------------------------------------------------------------
// Host launcher. Workspace layout (ushort elements):
//   W16   : 1024*1024                ( 2 MB)
//   E16   : 32*512*1024              (32 MB)
//   P16   : 32*512*1024              (32 MB)
//   WD16  : 32*2048*1024             (128 MB)  row-major words bf16
//   WD16T : 32*1024*2048             (128 MB)  transposed words bf16
// ---------------------------------------------------------------------------
extern "C" void kernel_launch(void* const* d_in, const int* in_sizes, int n_in,
                              void* d_out, int out_size, void* d_ws, size_t ws_size,
                              hipStream_t stream) {
  const float*         ems   = (const float*)d_in[0];
  const float*         words = (const float*)d_in[1];
  const unsigned char* msk   = (const unsigned char*)d_in[2];
  const float*         wW    = (const float*)d_in[3];
  const float*         wB    = (const float*)d_in[4];
  float*               out   = (float*)d_out;

  unsigned short* W16   = (unsigned short*)d_ws;
  unsigned short* E16   = W16  + (size_t)D_DIM * D_DIM;
  unsigned short* P16   = E16  + (size_t)B_DIM * E_DIM * D_DIM;
  unsigned short* WD16  = P16  + (size_t)B_DIM * E_DIM * D_DIM;
  unsigned short* WD16T = WD16 + (size_t)B_DIM * W_N * D_DIM;

  int n4;
  n4 = (D_DIM * D_DIM) / 4;
  cvt_f32_to_bf16<<<(n4 + 255) / 256, 256, 0, stream>>>(wW, W16, n4);
  n4 = (B_DIM * E_DIM * D_DIM) / 4;
  cvt_f32_to_bf16<<<(n4 + 255) / 256, 256, 0, stream>>>(ems, E16, n4);
  n4 = (B_DIM * W_N * D_DIM) / 4;
  cvt_f32_to_bf16<<<(n4 + 255) / 256, 256, 0, stream>>>(words, WD16, n4);

  dim3 tg(D_DIM / 32, W_N / 32, B_DIM);
  transpose_cvt_kernel<<<tg, dim3(32, 8), 0, stream>>>(words, WD16T);

  proj_tanh_kernel<<<(B_DIM * E_DIM) / 16, 256, 0, stream>>>(E16, W16, wB, P16);
  attn_kernel<<<B_DIM * (E_DIM / 16), 256, 0, stream>>>(P16, WD16, WD16T, msk, out);
}
